// NormalNet_85134841741848
// MI455X (gfx1250) — matrix operand order repaired and checked by reading, not compiled, not run on hardware
//
#include <hip/hip_runtime.h>
#include <hip/hip_bf16.h>
#include <math.h>

typedef __attribute__((ext_vector_type(16))) _Float16 v16h;
typedef __attribute__((ext_vector_type(2)))  _Float16 v2h;
typedef __attribute__((ext_vector_type(8)))  float    v8f;

#define NN 30000
#define NE 480000
#define NG 64

// ---------------------------------------------------------------------------
__global__ void fill_kernel(float* __restrict__ p, float v, long long n) {
    long long i = (long long)blockIdx.x * blockDim.x + threadIdx.x;
    if (i < n) p[i] = v;
}

// ---------------------------------------------------------------------------
// WMMA f16 GEMM: out[M,Nc] = act(X[M,K] @ W[K,Nc] + bias[Nc])
// blockDim = 128 (4 waves). Block tile: 64 rows x 64 cols; wave w owns rows
// [m0+16w, m0+16w+16) and multiplies its A fragment against the 4 B sub-tiles
// -> 4 v_wmma per wave per K-step. LDS tiles are kept in *fragment order* so
// each lane reads its v16h fragment as contiguous 32B (2x ds_load_b128), and
// staging writes packed k-pairs (v2h -> ds_store_b32, v_cvt_pk_f16_f32).
// Requires K%32==0, Nc%64==0. M guarded (clamped loads / guarded stores).
__global__ void wmma_gemm_bias(const float* __restrict__ X,
                               const float* __restrict__ W,
                               const float* __restrict__ bias,
                               float* __restrict__ out,
                               int M, int K, int Nc, int relu) {
    __shared__ v16h AsF[128];   // [wave][lane] fragment order
    __shared__ v16h BsF[128];   // [nsub][lane] fragment order

    const int tid  = threadIdx.x;
    const int lane = tid & 31;
    const int wv   = tid >> 5;
    const int half = lane >> 4;
    const int lm   = lane & 15;
    const int m0   = blockIdx.x * 64;
    const int nb0  = blockIdx.y * 64;

    v2h* Ah2 = (v2h*)AsF;
    v2h* Bh2 = (v2h*)BsF;

    v8f acc[4];
#pragma unroll
    for (int t = 0; t < 4; ++t) {
        float bb = bias[nb0 + t * 16 + lm];
#pragma unroll
        for (int r = 0; r < 8; ++r) acc[t][r] = bb;
    }

    for (int kb = 0; kb < K; kb += 32) {
        // ---- stage A (64 rows x 32 k): float4 load -> 2 packed b32 stores --
#pragma unroll
        for (int cc = 0; cc < 4; ++cc) {
            int idx = cc * 128 + tid;           // 0..511 groups of 4 k
            int r   = idx >> 3;                 // local row 0..63
            int c4  = (idx & 7) * 4;            // local k 0,4,...,28
            int rg  = m0 + r; if (rg >= M) rg = M - 1;
            const float4 f = *(const float4*)(X + (long long)rg * K + kb + c4);
            int w = r >> 4, mloc = r & 15;
            int hf   = (c4 & 15) >> 3;                    // same for all 4 k
            int i0   = ((c4 >> 4) << 3) + (c4 & 7);       // fragment index of c4
            int slot = (w * 32 + hf * 16 + mloc) * 16;    // half-index base
            v2h p0; p0[0] = (_Float16)f.x; p0[1] = (_Float16)f.y;
            v2h p1; p1[0] = (_Float16)f.z; p1[1] = (_Float16)f.w;
            Ah2[(slot + i0) >> 1]     = p0;
            Ah2[(slot + i0 + 2) >> 1] = p1;
        }
        // ---- stage B (32 k x 64 n): 2 coalesced rows -> 1 packed b32 store -
#pragma unroll
        for (int cc = 0; cc < 8; ++cc) {
            int idx = cc * 128 + tid;           // 0..1023 (k-pairs x n)
            int kp  = idx >> 6;                 // k-pair 0..15 -> kk = 2*kp
            int nn  = idx & 63;
            int kk  = kp * 2;
            float v0 = W[(long long)(kb + kk) * Nc + nb0 + nn];
            float v1 = W[(long long)(kb + kk + 1) * Nc + nb0 + nn];
            int slot = ((nn >> 4) * 32 + (((kk >> 3) & 1) << 4) + (nn & 15)) * 16;
            v2h p; p[0] = (_Float16)v0; p[1] = (_Float16)v1;
            Bh2[(slot + ((kk >> 4) << 3) + (kk & 7)) >> 1] = p;
        }
        __syncthreads();

        v16h a = AsF[wv * 32 + lane];
#pragma unroll
        for (int t = 0; t < 4; ++t) {
            v16h b = BsF[t * 32 + lane];
            acc[t] = __builtin_amdgcn_wmma_f32_16x16x32_f16(
                         false, a, false, b, (short)0, acc[t], false, false);
        }
        __syncthreads();
    }

#pragma unroll
    for (int t = 0; t < 4; ++t) {
        int col = nb0 + t * 16 + lm;
#pragma unroll
        for (int r = 0; r < 8; ++r) {
            int row = m0 + wv * 16 + half * 8 + r;
            float v = acc[t][r];
            if (relu) v = fmaxf(v, 0.f);
            if (row < M) out[(long long)row * Nc + col] = v;
        }
    }
}

// ---------------------------------------------------------------------------
// per-(edge,head) attention logits: a = <q_dst, k_src + ea*ew> / sqrt(dh)
__global__ void edge_logits(const int* __restrict__ src, const int* __restrict__ dst,
                            const float* __restrict__ Q, const float* __restrict__ Kf,
                            const float* __restrict__ ea, const float* __restrict__ ew,
                            float* __restrict__ a, int H, int dh, int o) {
    long long t = (long long)blockIdx.x * blockDim.x + threadIdx.x;
    if (t >= (long long)NE * H) return;
    int e = (int)(t / H), h = (int)(t % H);
    int s = src[e], d = dst[e];
    float eav = ea[e];
    const float4* qp = (const float4*)(Q  + (long long)d * o + h * dh);
    const float4* kp = (const float4*)(Kf + (long long)s * o + h * dh);
    const float4* ep = (const float4*)(ew + h * dh);
    float sum = 0.f;
    for (int j = 0; j < dh / 4; ++j) {
        float4 q4 = qp[j], k4 = kp[j], e4 = ep[j];
        sum += q4.x * (k4.x + eav * e4.x) + q4.y * (k4.y + eav * e4.y)
             + q4.z * (k4.z + eav * e4.z) + q4.w * (k4.w + eav * e4.w);
    }
    a[t] = sum * rsqrtf((float)dh);
}

__device__ inline void atomicMaxF(float* addr, float val) {
    int old = __float_as_int(*addr);
    while (__int_as_float(old) < val) {
        int assumed = old;
        old = atomicCAS((int*)addr, assumed, __float_as_int(val));
        if (old == assumed) break;
    }
}

__global__ void seg_max_kernel(const int* __restrict__ dst, const float* __restrict__ a,
                               float* __restrict__ amax, int H) {
    long long t = (long long)blockIdx.x * blockDim.x + threadIdx.x;
    if (t >= (long long)NE * H) return;
    int e = (int)(t / H), h = (int)(t % H);
    atomicMaxF(&amax[(long long)dst[e] * H + h], a[t]);
}

__global__ void edge_exp_kernel(const int* __restrict__ dst, float* __restrict__ a,
                                const float* __restrict__ amax, float* __restrict__ den, int H) {
    long long t = (long long)blockIdx.x * blockDim.x + threadIdx.x;
    if (t >= (long long)NE * H) return;
    int e = (int)(t / H), h = (int)(t % H);
    float ex = expf(a[t] - amax[(long long)dst[e] * H + h]);
    a[t] = ex;
    atomicAdd(&den[(long long)dst[e] * H + h], ex);
}

// per-(edge, 4 channels) scatter: agg[dst,c] += alpha * (V[src,c] + ea*ew[c])
__global__ void edge_scatter_kernel(const int* __restrict__ src, const int* __restrict__ dst,
                                    const float* __restrict__ V, const float* __restrict__ ea,
                                    const float* __restrict__ ew, const float* __restrict__ ex,
                                    const float* __restrict__ den, float* __restrict__ agg,
                                    int H, int dh, int o) {
    int oq = o >> 2;
    long long t = (long long)blockIdx.x * blockDim.x + threadIdx.x;
    if (t >= (long long)NE * oq) return;
    int e  = (int)(t / oq), cq = (int)(t % oq);
    int c4 = cq * 4, h = c4 / dh;
    int s = src[e], d = dst[e];
    float alpha = ex[(long long)e * H + h] / den[(long long)d * H + h];
    float eav = ea[e];
    float4 v4 = *(const float4*)(V + (long long)s * o + c4);
    float4 w4 = *(const float4*)(ew + c4);
    float* ap = agg + (long long)d * o + c4;
    atomicAdd(ap + 0, (v4.x + eav * w4.x) * alpha);
    atomicAdd(ap + 1, (v4.y + eav * w4.y) * alpha);
    atomicAdd(ap + 2, (v4.z + eav * w4.z) * alpha);
    atomicAdd(ap + 3, (v4.w + eav * w4.w) * alpha);
}

__global__ void add_skip_act(const float* __restrict__ agg, const float* __restrict__ S,
                             float* __restrict__ out, long long n, int relu) {
    long long t = (long long)blockIdx.x * blockDim.x + threadIdx.x;
    if (t >= n) return;
    float v = agg[t] + S[t];
    if (relu) v = fmaxf(v, 0.f);
    out[t] = v;
}

// ---------------------------------------------------------------------------
__global__ void pool_kernel(const float* __restrict__ h, const int* __restrict__ batch,
                            float* __restrict__ psum, float* __restrict__ cnt) {
    long long t = (long long)blockIdx.x * blockDim.x + threadIdx.x;
    if (t >= (long long)NN * 128) return;
    int n = (int)(t >> 7), c = (int)(t & 127);
    int g = batch[n];
    atomicAdd(&psum[g * 128 + c], h[t]);
    if (c == 0) atomicAdd(&cnt[g], 1.0f);
}

__global__ void final_linear(const float* __restrict__ psum, const float* __restrict__ cnt,
                             const float* __restrict__ lw, const float* __restrict__ lb,
                             float* __restrict__ out) {
    int t = blockIdx.x * blockDim.x + threadIdx.x;
    if (t >= NG * 13) return;
    int g = t / 13, oc = t % 13;
    float c = cnt[g]; c = c > 1.f ? c : 1.f;
    float sum = lb[oc];
    for (int j = 0; j < 128; ++j) sum += (psum[g * 128 + j] / c) * lw[j * 13 + oc];
    out[t] = sum;
}

// ---------------------------------------------------------------------------
extern "C" void kernel_launch(void* const* d_in, const int* in_sizes, int n_in,
                              void* d_out, int out_size, void* d_ws, size_t ws_size,
                              hipStream_t stream) {
    const float* x     = (const float*)d_in[0];
    const int*   ei    = (const int*)d_in[1];      // [2,E]
    const float* ea    = (const float*)d_in[2];    // [E,1]
    const int*   batch = (const int*)d_in[3];

    const int* src = ei;
    const int* dst = ei + NE;

    float* ws = (float*)d_ws;
    size_t NSZ = (size_t)NN;
    float* Qb  = ws;                   // N*512
    float* Kb  = Qb  + NSZ * 512;
    float* Vb  = Kb  + NSZ * 512;
    float* Sb  = Vb  + NSZ * 512;
    float* Ag  = Sb  + NSZ * 512;
    float* Hb  = Ag  + NSZ * 512;
    float* Hb2 = Hb  + NSZ * 512;
    float* Eb  = Hb2 + NSZ * 512;      // E*8
    float* Mx  = Eb  + (size_t)NE * 8; // N*8
    float* Dn  = Mx  + NSZ * 8;        // N*8
    float* Ps  = Dn  + NSZ * 8;        // G*128
    float* Ct  = Ps  + (size_t)NG * 128;

    auto cdiv = [](long long a, long long b) { return (unsigned)((a + b - 1) / b); };

    auto run_layer = [&](const float* Xin, int K, int H, int dh,
                         const float* qw, const float* qb, const float* kw, const float* kb2,
                         const float* vw, const float* vb, const float* ewp,
                         const float* sw, const float* sb2, float* Hout, int relu) {
        int o = H * dh;
        dim3 gg(cdiv(NN, 64), o / 64);
        wmma_gemm_bias<<<gg, 128, 0, stream>>>(Xin, qw, qb,  Qb, NN, K, o, 0);
        wmma_gemm_bias<<<gg, 128, 0, stream>>>(Xin, kw, kb2, Kb, NN, K, o, 0);
        wmma_gemm_bias<<<gg, 128, 0, stream>>>(Xin, vw, vb,  Vb, NN, K, o, 0);
        wmma_gemm_bias<<<gg, 128, 0, stream>>>(Xin, sw, sb2, Sb, NN, K, o, 0);

        long long nNH = (long long)NN * H;
        long long nEH = (long long)NE * H;
        long long nNo = (long long)NN * o;
        long long nEq = (long long)NE * (o / 4);
        fill_kernel<<<cdiv(nNH, 256), 256, 0, stream>>>(Mx, -1e30f, nNH);
        fill_kernel<<<cdiv(nNH, 256), 256, 0, stream>>>(Dn, 0.f, nNH);
        fill_kernel<<<cdiv(nNo, 256), 256, 0, stream>>>(Ag, 0.f, nNo);

        edge_logits<<<cdiv(nEH, 256), 256, 0, stream>>>(src, dst, Qb, Kb, ea, ewp, Eb, H, dh, o);
        seg_max_kernel<<<cdiv(nEH, 256), 256, 0, stream>>>(dst, Eb, Mx, H);
        edge_exp_kernel<<<cdiv(nEH, 256), 256, 0, stream>>>(dst, Eb, Mx, Dn, H);
        edge_scatter_kernel<<<cdiv(nEq, 256), 256, 0, stream>>>(src, dst, Vb, ea, ewp, Eb, Dn, Ag, H, dh, o);
        add_skip_act<<<cdiv(nNo, 256), 256, 0, stream>>>(Ag, Sb, Hout, nNo, relu);
    };

    // layer 1: 256 -> 8x64 (512), relu
    run_layer(x, 256, 8, 64,
              (const float*)d_in[4],  (const float*)d_in[5],
              (const float*)d_in[6],  (const float*)d_in[7],
              (const float*)d_in[8],  (const float*)d_in[9],
              (const float*)d_in[10],
              (const float*)d_in[11], (const float*)d_in[12], Hb, 1);
    // layer 2: 512 -> 4x64 (256), relu
    run_layer(Hb, 512, 4, 64,
              (const float*)d_in[13], (const float*)d_in[14],
              (const float*)d_in[15], (const float*)d_in[16],
              (const float*)d_in[17], (const float*)d_in[18],
              (const float*)d_in[19],
              (const float*)d_in[20], (const float*)d_in[21], Hb2, 1);
    // layer 3: 256 -> 1x128 (128), no relu
    run_layer(Hb2, 256, 1, 128,
              (const float*)d_in[22], (const float*)d_in[23],
              (const float*)d_in[24], (const float*)d_in[25],
              (const float*)d_in[26], (const float*)d_in[27],
              (const float*)d_in[28],
              (const float*)d_in[29], (const float*)d_in[30], Hb, 0);

    // mean pool + final linear
    fill_kernel<<<cdiv(NG * 128, 256), 256, 0, stream>>>(Ps, 0.f, NG * 128);
    fill_kernel<<<1, 64, 0, stream>>>(Ct, 0.f, NG);
    pool_kernel<<<cdiv((long long)NN * 128, 256), 256, 0, stream>>>(Hb, batch, Ps, Ct);
    final_linear<<<cdiv(NG * 13, 256), 256, 0, stream>>>(Ps, Ct,
                 (const float*)d_in[31], (const float*)d_in[32], (float*)d_out);
}
